// SwinMLPBlockTriton_19069654794881
// MI455X (gfx1250) — matrix operand order repaired and checked by reading, not compiled, not run on hardware
//
#include <hip/hip_runtime.h>
#include <hip/hip_bf16.h>
#include <math.h>

// ---------------- CDNA5 WMMA types ----------------
typedef __attribute__((ext_vector_type(16))) __bf16 v16bf;
typedef __attribute__((ext_vector_type(8)))  float  v8f;

#define DIM     512
#define NHEADS  16
#define HDIM    32
#define HIDDEN  2048
#define MT      64      // tokens per workgroup
#define CHUNK   128     // hidden-dim chunk
#define EPSV    1e-5f

union FragBits {
    uint4 q[2];
    v16bf v;
};

__device__ __forceinline__ unsigned f2bf(float f) {
    union { float f; unsigned u; } c; c.f = f;
    unsigned u = c.u;
    u += 0x7FFFu + ((u >> 16) & 1u);       // round-to-nearest-even
    return u >> 16;
}
__device__ __forceinline__ unsigned pack2bf(float lo, float hi) {
    return f2bf(lo) | (f2bf(hi) << 16);
}

__device__ __forceinline__ float gelu_tanh(float x) {
    const float k0 = 0.7978845608028654f;  // sqrt(2/pi)
    const float k1 = 0.044715f;
    return 0.5f * x * (1.0f + tanhf(k0 * (x + k1 * x * x * x)));
}

// 16x32 bf16 A-fragment from GLOBAL feature-major weights (row m contiguous in k).
// ISA 7.12.2 A layout: lanes 0-15 (m=lane) hold K {0..7,16..23}; lanes 16-31 K {8..15,24..31}.
__device__ __forceinline__ v16bf load_a_glb(const unsigned short* __restrict__ wT,
                                            int rowBase, int stride, int kBase) {
    const int lane = threadIdx.x & 31;
    const int hi   = lane >> 4;
    const unsigned short* p = wT + (size_t)(rowBase + (lane & 15)) * stride + kBase;
    FragBits f;
    f.q[0] = *(const uint4*)(p + 8 * hi);
    f.q[1] = *(const uint4*)(p + 16 + 8 * hi);
    return f.v;
}

// 32x16 bf16 B-fragment from LDS token-major activations:
// lane = token column n; K {0..7,16..23} for lanes 0-15, K {8..15,24..31} for lanes 16-31.
__device__ __forceinline__ v16bf load_b_lds(const unsigned short* p0,
                                            int tokBase, int stride, int kBase) {
    const int lane = threadIdx.x & 31;
    const unsigned short* p =
        p0 + (tokBase + (lane & 15)) * stride + kBase + 8 * (lane >> 4);
    FragBits f;
    f.q[0] = *(const uint4*)(p);
    f.q[1] = *(const uint4*)(p + 16);
    return f.v;
}

// ---------------- weight prep: fp32 -> bf16, feature-major transpose ----------------
__global__ void prep_weights_kernel(const float* __restrict__ w1,   // [512][2048]
                                    const float* __restrict__ w2,   // [2048][512]
                                    const float* __restrict__ wsp,  // [16][32][32]
                                    unsigned short* __restrict__ w1t,   // [2048 h][512 k]
                                    unsigned short* __restrict__ w2t,   // [512 c][2048 h]
                                    unsigned short* __restrict__ wspT)  // [16][32 e][32 d]
{
    int idx = blockIdx.x * blockDim.x + threadIdx.x;
    if (idx < HIDDEN * DIM) {                        // w1t[n][k] = w1[k][n]
        int n = idx >> 9, k = idx & 511;
        w1t[idx] = (unsigned short)f2bf(w1[k * HIDDEN + n]);
    } else if (idx < 2 * HIDDEN * DIM) {             // w2t[c][h] = w2[h][c]
        int i = idx - HIDDEN * DIM;
        int c = i >> 11, h = i & 2047;
        w2t[i] = (unsigned short)f2bf(w2[h * DIM + c]);
    } else if (idx < 2 * HIDDEN * DIM + NHEADS * HDIM * HDIM) {
        int i = idx - 2 * HIDDEN * DIM;              // wspT[h][e][d] = wsp[h][d][e]
        int h = i >> 10, rr = i & 1023, e = rr >> 5, d = rr & 31;
        wspT[i] = (unsigned short)f2bf(wsp[(h * HDIM + d) * HDIM + e]);
    }
}

// ---------------- fully fused block kernel ----------------
// 256 threads = 8 wave32; 64 tokens per workgroup end-to-end.
// All WMMAs: A = feature-major weights (global), B = token-major activations (LDS),
// so D puts 8 consecutive feature columns of one token row in a lane -> all
// epilogues are float4 / packed-bf16 vector ops. Wave w owns a disjoint
// feature slice in every phase (no duplicated weight fetches in a workgroup).
__global__ __launch_bounds__(256, 1)
void swin_mlp_block_kernel(const float* __restrict__ x,
                           const float* __restrict__ g1, const float* __restrict__ be1,
                           const float* __restrict__ g2, const float* __restrict__ be2,
                           const float* __restrict__ b1, const float* __restrict__ b2,
                           const unsigned short* __restrict__ w1t,
                           const unsigned short* __restrict__ w2t,
                           const unsigned short* __restrict__ wspT,
                           float* __restrict__ out)
{
    extern __shared__ char smem[];
    float*          xbuf = (float*)smem;                                  // [64][512] x, then x2
    unsigned short* abuf = (unsigned short*)(smem + 131072);              // [64][512] bf16 LN out
    unsigned short* act  = (unsigned short*)(smem + 131072 + 65536);      // [64][128] bf16
    float*          red  = (float*)(smem + 131072 + 65536 + 16384);       // [64][4][2]
    float*          lnp  = (float*)(smem + 131072 + 65536 + 16384 + 2048);// [64][2]

    const int tid  = threadIdx.x;
    const int w    = tid >> 5;   // wave id 0..7
    const int lane = tid & 31;
    const int hi   = lane >> 4;
    const int ln15 = lane & 15;
    const size_t rowG = (size_t)blockIdx.x * MT;

    const int r    = tid & 63;   // token row for LN passes
    const int part = tid >> 6;   // 128-column quarter

    // ---- Stage x tile (contiguous 128KB) into LDS via CDNA5 async-to-LDS DMA ----
    {
        unsigned lds0 = (unsigned)(size_t)xbuf + (unsigned)tid * 16u;
        unsigned long long g0 =
            (unsigned long long)(const void*)(x + rowG * DIM) + (unsigned long long)tid * 16ull;
        #pragma unroll 8
        for (int i = 0; i < 32; ++i) {   // 256 thr * 16B * 32 = 128KB
            unsigned lo = lds0 + (unsigned)i * 4096u;
            unsigned long long ga = g0 + (unsigned long long)i * 4096ull;
            asm volatile("global_load_async_to_lds_b128 %0, %1, off"
                         :: "v"(lo), "v"(ga) : "memory");
        }
        asm volatile("s_wait_asynccnt 0x0" ::: "memory");
    }
    __syncthreads();

    // ---- LN1 stats (from LDS, float4) ----
    {
        const float* xl = xbuf + r * DIM + part * 128;
        float s = 0.f, s2 = 0.f;
        #pragma unroll 8
        for (int c = 0; c < 128; c += 4) {
            float4 v = *(const float4*)(xl + c);
            s  += v.x + v.y + v.z + v.w;
            s2 += v.x * v.x + v.y * v.y + v.z * v.z + v.w * v.w;
        }
        red[(r * 4 + part) * 2 + 0] = s;
        red[(r * 4 + part) * 2 + 1] = s2;
    }
    __syncthreads();
    if (tid < 64) {
        float s = 0.f, s2 = 0.f;
        #pragma unroll
        for (int p = 0; p < 4; ++p) {
            s  += red[(tid * 4 + p) * 2 + 0];
            s2 += red[(tid * 4 + p) * 2 + 1];
        }
        float mu  = s * (1.0f / DIM);
        float var = s2 * (1.0f / DIM) - mu * mu;
        lnp[tid * 2 + 0] = mu;
        lnp[tid * 2 + 1] = rsqrtf(var + EPSV);
    }
    __syncthreads();
    {   // LN1 apply -> abuf (bf16, packed 4-wide stores)
        float mu = lnp[r * 2 + 0], rs = lnp[r * 2 + 1];
        #pragma unroll 4
        for (int c = 0; c < 128; c += 4) {
            int cc = part * 128 + c;
            float4 xv = *(const float4*)(xbuf + r * DIM + cc);
            float4 gv = *(const float4*)(g1 + cc);
            float4 bv = *(const float4*)(be1 + cc);
            uint2 pk;
            pk.x = pack2bf((xv.x - mu) * rs * gv.x + bv.x, (xv.y - mu) * rs * gv.y + bv.y);
            pk.y = pack2bf((xv.z - mu) * rs * gv.z + bv.z, (xv.w - mu) * rs * gv.w + bv.w);
            *(uint2*)(abuf + r * DIM + cc) = pk;
        }
    }
    __syncthreads();

    // ---- Spatial mixing: D = wsp_head^T(16e x 32d) @ LN1(32d x 16tok); x2 = x + D ----
    // wave w -> heads {2w, 2w+1}; all 4 token tiles.
    #pragma unroll
    for (int hh = 0; hh < 2; ++hh) {
        const int h = 2 * w + hh;
        v16bf a0 = load_a_glb(wspT + h * HDIM * HDIM, 0,  HDIM, 0);  // e-tile 0
        v16bf a1 = load_a_glb(wspT + h * HDIM * HDIM, 16, HDIM, 0);  // e-tile 1
        #pragma unroll
        for (int mt = 0; mt < 4; ++mt) {
            v16bf b = load_b_lds(abuf, mt * 16, DIM, h * HDIM);
            v8f d0 = {}, d1 = {};
            d0 = __builtin_amdgcn_wmma_f32_16x16x32_bf16(false, a0, false, b,
                                                         (short)0, d0, false, false);
            d1 = __builtin_amdgcn_wmma_f32_16x16x32_bf16(false, a1, false, b,
                                                         (short)0, d1, false, false);
            // lane: token = mt*16 + ln15; 8 consecutive e columns starting at 8*hi
            float* xp = xbuf + (mt * 16 + ln15) * DIM + h * HDIM + 8 * hi;
            float4 u0 = *(float4*)(xp);       float4 u1 = *(float4*)(xp + 4);
            float4 u2 = *(float4*)(xp + 16);  float4 u3 = *(float4*)(xp + 20);
            u0.x += d0[0]; u0.y += d0[1]; u0.z += d0[2]; u0.w += d0[3];
            u1.x += d0[4]; u1.y += d0[5]; u1.z += d0[6]; u1.w += d0[7];
            u2.x += d1[0]; u2.y += d1[1]; u2.z += d1[2]; u2.w += d1[3];
            u3.x += d1[4]; u3.y += d1[5]; u3.z += d1[6]; u3.w += d1[7];
            *(float4*)(xp)      = u0;  *(float4*)(xp + 4)  = u1;
            *(float4*)(xp + 16) = u2;  *(float4*)(xp + 20) = u3;
        }
    }
    __syncthreads();

    // ---- LN2 stats over x2 (float4) ----
    {
        const float* xl = xbuf + r * DIM + part * 128;
        float s = 0.f, s2 = 0.f;
        #pragma unroll 8
        for (int c = 0; c < 128; c += 4) {
            float4 v = *(const float4*)(xl + c);
            s  += v.x + v.y + v.z + v.w;
            s2 += v.x * v.x + v.y * v.y + v.z * v.z + v.w * v.w;
        }
        red[(r * 4 + part) * 2 + 0] = s;
        red[(r * 4 + part) * 2 + 1] = s2;
    }
    __syncthreads();
    if (tid < 64) {
        float s = 0.f, s2 = 0.f;
        #pragma unroll
        for (int p = 0; p < 4; ++p) {
            s  += red[(tid * 4 + p) * 2 + 0];
            s2 += red[(tid * 4 + p) * 2 + 1];
        }
        float mu  = s * (1.0f / DIM);
        float var = s2 * (1.0f / DIM) - mu * mu;
        lnp[tid * 2 + 0] = mu;
        lnp[tid * 2 + 1] = rsqrtf(var + EPSV);
    }
    __syncthreads();
    {   // LN2 apply -> abuf
        float mu = lnp[r * 2 + 0], rs = lnp[r * 2 + 1];
        #pragma unroll 4
        for (int c = 0; c < 128; c += 4) {
            int cc = part * 128 + c;
            float4 xv = *(const float4*)(xbuf + r * DIM + cc);
            float4 gv = *(const float4*)(g2 + cc);
            float4 bv = *(const float4*)(be2 + cc);
            uint2 pk;
            pk.x = pack2bf((xv.x - mu) * rs * gv.x + bv.x, (xv.y - mu) * rs * gv.y + bv.y);
            pk.y = pack2bf((xv.z - mu) * rs * gv.z + bv.z, (xv.w - mu) * rs * gv.w + bv.w);
            *(uint2*)(abuf + r * DIM + cc) = pk;
        }
    }
    __syncthreads();

    // ---- MLP: 16 hidden chunks of 128 ----
    v8f zero = {};
    v8f acc2[16];                         // [ct][mt]: wave w owns out channels [w*64, +64)
    #pragma unroll
    for (int i = 0; i < 16; ++i) acc2[i] = zero;

    #pragma unroll 1
    for (int chunk = 0; chunk < HIDDEN / CHUNK; ++chunk) {
        if (chunk + 1 < HIDDEN / CHUNK)   // global_prefetch next W1 slab
            __builtin_prefetch(w1t + (size_t)((chunk + 1) * CHUNK + w * 16) * DIM, 0, 1);

        // GEMM1: D = W1^T(16h x 32k) @ LN2(32k x 16tok); wave w -> hidden rows chunk*128+w*16
        const int hbase = chunk * CHUNK + w * 16;
        v8f a1[4];
        #pragma unroll
        for (int t = 0; t < 4; ++t) a1[t] = zero;

        #pragma unroll 4
        for (int ks = 0; ks < 16; ++ks) {              // K = 512 in steps of 32
            v16bf a = load_a_glb(w1t, hbase, DIM, ks * 32);
            #pragma unroll
            for (int mt = 0; mt < 4; ++mt) {
                v16bf b = load_b_lds(abuf, mt * 16, DIM, ks * 32);
                a1[mt] = __builtin_amdgcn_wmma_f32_16x16x32_bf16(false, a, false, b,
                                                                 (short)0, a1[mt], false, false);
            }
        }
        {   // bias + gelu + packed act store: lane = token, 8 consecutive hidden cols
            float4 bb0 = *(const float4*)(b1 + hbase + 8 * hi);
            float4 bb1 = *(const float4*)(b1 + hbase + 8 * hi + 4);
            #pragma unroll
            for (int mt = 0; mt < 4; ++mt) {
                uint4 pk;
                pk.x = pack2bf(gelu_tanh(a1[mt][0] + bb0.x), gelu_tanh(a1[mt][1] + bb0.y));
                pk.y = pack2bf(gelu_tanh(a1[mt][2] + bb0.z), gelu_tanh(a1[mt][3] + bb0.w));
                pk.z = pack2bf(gelu_tanh(a1[mt][4] + bb1.x), gelu_tanh(a1[mt][5] + bb1.y));
                pk.w = pack2bf(gelu_tanh(a1[mt][6] + bb1.z), gelu_tanh(a1[mt][7] + bb1.w));
                *(uint4*)(act + (mt * 16 + ln15) * CHUNK + w * 16 + 8 * hi) = pk;
            }
        }
        __syncthreads();   // act tile complete

        // GEMM2: D = W2^T(16c x 32k) @ act(32k x 16tok); wave w -> out channels [w*64,+64)
        #pragma unroll 2
        for (int k2 = 0; k2 < 4; ++k2) {               // 128 in steps of 32
            v16bf a[4];
            #pragma unroll
            for (int ct = 0; ct < 4; ++ct)
                a[ct] = load_a_glb(w2t, w * 64 + ct * 16, HIDDEN,
                                   chunk * CHUNK + k2 * 32);
            #pragma unroll
            for (int mt = 0; mt < 4; ++mt) {
                v16bf b = load_b_lds(act, mt * 16, CHUNK, k2 * 32);
                #pragma unroll
                for (int ct = 0; ct < 4; ++ct) {
                    acc2[ct * 4 + mt] =
                        __builtin_amdgcn_wmma_f32_16x16x32_bf16(false, a[ct], false, b,
                                                                (short)0, acc2[ct * 4 + mt],
                                                                false, false);
                }
            }
        }
        __syncthreads();   // safe to overwrite act next chunk
    }

    // ---- Epilogue: out = x2 + y + b2  (32B contiguous per lane) ----
    #pragma unroll
    for (int ct = 0; ct < 4; ++ct) {
        const int col0 = w * 64 + ct * 16 + 8 * hi;
        float4 c0 = *(const float4*)(b2 + col0);
        float4 c1 = *(const float4*)(b2 + col0 + 4);
        #pragma unroll
        for (int mt = 0; mt < 4; ++mt) {
            const int tok = mt * 16 + ln15;
            const float* xp = xbuf + tok * DIM + col0;
            float4 x0 = *(const float4*)(xp);
            float4 x1 = *(const float4*)(xp + 4);
            const v8f& y = acc2[ct * 4 + mt];
            float4 o0, o1;
            o0.x = x0.x + y[0] + c0.x; o0.y = x0.y + y[1] + c0.y;
            o0.z = x0.z + y[2] + c0.z; o0.w = x0.w + y[3] + c0.w;
            o1.x = x1.x + y[4] + c1.x; o1.y = x1.y + y[5] + c1.y;
            o1.z = x1.z + y[6] + c1.z; o1.w = x1.w + y[7] + c1.w;
            float* op = out + (rowG + tok) * DIM + col0;
            *(float4*)(op)     = o0;
            *(float4*)(op + 4) = o1;
        }
    }
}

// ---------------- host entry ----------------
extern "C" void kernel_launch(void* const* d_in, const int* in_sizes, int n_in,
                              void* d_out, int out_size, void* d_ws, size_t ws_size,
                              hipStream_t stream) {
    const float* x   = (const float*)d_in[0];
    const float* g1  = (const float*)d_in[1];
    const float* be1 = (const float*)d_in[2];
    const float* wsp = (const float*)d_in[3];
    const float* g2  = (const float*)d_in[4];
    const float* be2 = (const float*)d_in[5];
    const float* w1  = (const float*)d_in[6];
    const float* b1  = (const float*)d_in[7];
    const float* w2  = (const float*)d_in[8];
    const float* b2  = (const float*)d_in[9];
    float* out = (float*)d_out;

    unsigned short* w1t  = (unsigned short*)d_ws;              // 2048*512 bf16
    unsigned short* w2t  = w1t + HIDDEN * DIM;                 // 512*2048 bf16
    unsigned short* wspT = w2t + DIM * HIDDEN;                 // 16*32*32 bf16

    const int prepN = 2 * HIDDEN * DIM + NHEADS * HDIM * HDIM;
    prep_weights_kernel<<<(prepN + 255) / 256, 256, 0, stream>>>(w1, w2, wsp, w1t, w2t, wspT);

    const int tokens  = 32 * 3136;        // 100,352
    const int nblocks = tokens / MT;      // 1,568
    const size_t smem = 131072 + 65536 + 16384 + 2048 + 512;  // 215,552 B (<320KB/WG)
    swin_mlp_block_kernel<<<nblocks, 256, smem, stream>>>(
        x, g1, be1, g2, be2, b1, b2, w1t, w2t, wspT, out);
}
